// DeltaConvGRU_83425444757870
// MI455X (gfx1250) — matrix operand order repaired and checked by reading, not compiled, not run on hardware
//
#include <hip/hip_runtime.h>
#include <cstdint>
#include <cstddef>

// ---------------- problem constants ----------------
#define B_    8
#define CIN   32
#define CHID  64
#define H_    180
#define W_    240
#define TILE_H 6
#define TILE_W 16
#define PATCH_H 8     // TILE_H + 2
#define PATCH_W 18    // TILE_W + 2
#define PCH    96     // stacked channels (32 delta_x + 64 delta_h)
#define NPIX   (TILE_H*TILE_W)         // 96
#define SLICE  (B_*CHID*H_*W_)         // 22,118,400 elements per output tensor

// weight-prep offsets (ushort elements) in d_ws, layout [kyx][n][c] bf16
#define WOFS_U  0
#define WOFS_R  55296          // 9*64*96
#define WOFS_CX 110592         // + 9*64*96
#define WOFS_CH 129024         // + 9*64*32
#define WTOT    165888         // + 9*64*64

typedef __attribute__((ext_vector_type(16))) __bf16 v16bf;
typedef __attribute__((ext_vector_type(8)))  float  v8f;

__device__ __forceinline__ unsigned short f2bf(float f) {
  union { float f; uint32_t u; } c; c.f = f;
  uint32_t u = c.u;
  return (unsigned short)((u + 0x7FFFu + ((u >> 16) & 1u)) >> 16); // RNE
}

__device__ __forceinline__ v16bf make_frag(uint4 a, uint4 b) {
  union { v16bf v; uint4 q[2]; } u;
  u.q[0] = a; u.q[1] = b;
  return u.v;
}

// ---------------- weight prep: OIHW fp32 -> [kyx][n][c] bf16 ----------------
__global__ void prep_weights(const float* __restrict__ wu, const float* __restrict__ wr,
                             const float* __restrict__ wcx, const float* __restrict__ wch,
                             unsigned short* __restrict__ wp) {
  int g = blockIdx.x * 256 + threadIdx.x;
  if (g >= WTOT) return;
  int idx, Cg; const float* src;
  if (g < WOFS_R)        { idx = g;           Cg = 96; src = wu;  }
  else if (g < WOFS_CX)  { idx = g - WOFS_R;  Cg = 96; src = wr;  }
  else if (g < WOFS_CH)  { idx = g - WOFS_CX; Cg = 32; src = wcx; }
  else                   { idx = g - WOFS_CH; Cg = 64; src = wch; }
  int kyx = idx / (64 * Cg);
  int rem = idx - kyx * 64 * Cg;
  int n   = rem / Cg;
  int c   = rem - n * Cg;
  wp[g] = f2bf(src[(n * Cg + c) * 9 + kyx]);   // OIHW: ((n*Cg+c)*3+ky)*3+kx == (n*Cg+c)*9+kyx
}

// ---------------- main fused kernel ----------------
__global__ __launch_bounds__(256) void dgru_main(
    const float* __restrict__ x,   const float* __restrict__ ps,
    const float* __restrict__ pi,  const float* __restrict__ pps,
    const float* __restrict__ mu,  const float* __restrict__ mr,
    const float* __restrict__ mcx, const float* __restrict__ mch,
    const float* __restrict__ bu,  const float* __restrict__ br,
    const float* __restrict__ bcx, const float* __restrict__ bch,
    const unsigned short* __restrict__ wprep,
    float* __restrict__ out)
{
  __shared__ alignas(16) unsigned short patch[PATCH_H * PATCH_W * PCH]; // 27,648 B

  const int tx  = blockIdx.x;   // 0..14
  const int ty  = blockIdx.y;   // 0..29
  const int b   = blockIdx.z;   // 0..7
  const int tid = threadIdx.x;

  // ---- stage delta patch into LDS (bf16, channel-innermost) ----
  const int y0 = ty * TILE_H - 1;
  const int x0 = tx * TILE_W - 1;
#pragma unroll 1
  for (int i = 0; i < 54; ++i) {                 // 13824 / 256
    int e   = tid + i * 256;
    int ch  = e / (PATCH_H * PATCH_W);           // /144
    int rem = e - ch * (PATCH_H * PATCH_W);
    int row = rem / PATCH_W;
    int col = rem - row * PATCH_W;
    int yg = y0 + row, xg = x0 + col;
    float v = 0.f;
    if ((unsigned)yg < H_ && (unsigned)xg < W_) {
      if (ch < CIN) {
        int g = ((b * CIN + ch) * H_ + yg) * W_ + xg;
        v = x[g] - pi[g];
      } else {
        int g = ((b * CHID + (ch - CIN)) * H_ + yg) * W_ + xg;
        v = ps[g] - pps[g];
      }
    }
    patch[(row * PATCH_W + col) * PCH + ch] = f2bf(v);
  }
  __syncthreads();

  // ---- implicit-GEMM WMMA compute ----
  const int ln  = tid & 31;
  const int wid = tid >> 5;
  const int l15 = ln & 15;
  const int hi  = ln >> 4;           // lane half (K-interleave select)
  const int ntile = wid & 3;

  // waves 0-3 handle convs {u, c_x}; waves 4-7 handle {r, c_h}
#pragma unroll 1
  for (int uu = 0; uu < 2; ++uu) {
    const int conv = (uu == 0) ? ((wid < 4) ? 0 : 1) : ((wid < 4) ? 2 : 3);

    int Cg, nsteps, wofs, chbase0, slice;
    const float* mem; const float* bias;
    if (conv == 0)      { Cg = 96; nsteps = 27; wofs = WOFS_U;  mem = mu;  bias = bu;  slice = 1; chbase0 = 0;  }
    else if (conv == 1) { Cg = 96; nsteps = 27; wofs = WOFS_R;  mem = mr;  bias = br;  slice = 2; chbase0 = 0;  }
    else if (conv == 2) { Cg = 32; nsteps = 9;  wofs = WOFS_CX; mem = mcx; bias = bcx; slice = 3; chbase0 = 0;  }
    else                { Cg = 64; nsteps = 18; wofs = WOFS_CH; mem = mch; bias = bch; slice = 4; chbase0 = 32; }

    v8f acc[TILE_H];
#pragma unroll
    for (int t = 0; t < TILE_H; ++t)
#pragma unroll
      for (int e = 0; e < 8; ++e) acc[t][e] = 0.f;

    const int n = ntile * 16 + l15;              // output channel this lane owns (B/D layout)

#pragma unroll 1
    for (int s = 0; s < nsteps; ++s) {
      int kyx, c0;
      if (conv < 2)       { kyx = s / 3;  c0 = (s - kyx * 3) * 32; }
      else if (conv == 2) { kyx = s;      c0 = 0;                  }
      else                { kyx = s >> 1; c0 = (s & 1) * 32;       }

      // B fragment: K[8h..8h+7] and K[16+8h..16+8h+7] of prepped weights (L2-resident)
      const uint4* wp4 = (const uint4*)(wprep + wofs + (size_t)(kyx * 64 + n) * Cg + c0 + hi * 8);
      v16bf bfrag = make_frag(wp4[0], wp4[2]);

      const int ky = kyx / 3, kx = kyx - ky * 3;
      const int colb = l15 + kx;                 // patch column for this lane's M-row
      const int pcb  = chbase0 + c0;

#pragma unroll
      for (int t = 0; t < TILE_H; ++t) {
        const int off = ((t + ky) * PATCH_W + colb) * PCH + pcb + hi * 8; // ushort idx, 16B aligned
        const uint4* ap = (const uint4*)(patch + off);
        v16bf afrag = make_frag(ap[0], ap[2]);   // K[0..7|8..15] and K[16..23|24..31]
        acc[t] = __builtin_amdgcn_wmma_f32_16x16x32_bf16(
            false, afrag, false, bfrag, (short)0, acc[t], false, false);
      }
    }

    // epilogue: + bias + mem_*, store straight into output slice (lane holds 8 consecutive pixels)
    const float bv = bias[n];
#pragma unroll
    for (int t = 0; t < TILE_H; ++t) {
      const int y = ty * TILE_H + t;
      const size_t g = (((size_t)b * CHID + n) * H_ + y) * W_ + tx * TILE_W + hi * 8;
      const float4* mv = (const float4*)(mem + g);
      float4 m0 = mv[0], m1 = mv[1];
      float4 o0, o1;
      o0.x = acc[t][0] + bv + m0.x;  o0.y = acc[t][1] + bv + m0.y;
      o0.z = acc[t][2] + bv + m0.z;  o0.w = acc[t][3] + bv + m0.w;
      o1.x = acc[t][4] + bv + m1.x;  o1.y = acc[t][5] + bv + m1.y;
      o1.z = acc[t][6] + bv + m1.z;  o1.w = acc[t][7] + bv + m1.w;
      float4* op = (float4*)(out + (size_t)slice * SLICE + g);
      op[0] = o0; op[1] = o1;
    }
  }

  __threadfence();
  __syncthreads();

  // ---- fused gate elementwise: new_state from the four conv results ----
  const volatile float* uo  = out + 1 * (size_t)SLICE;
  const volatile float* ro  = out + 2 * (size_t)SLICE;
  const volatile float* cxo = out + 3 * (size_t)SLICE;
  const volatile float* cho = out + 4 * (size_t)SLICE;
#pragma unroll 1
  for (int i = 0; i < 24; ++i) {                 // 6144 / 256
    int idx = tid + i * 256;
    int ch  = idx / NPIX;
    int p   = idx - ch * NPIX;
    int y   = ty * TILE_H + (p >> 4);
    int xg  = tx * TILE_W + (p & 15);
    size_t g = (((size_t)b * CHID + ch) * H_ + y) * W_ + xg;
    float uv  = uo[g];
    float rv  = ro[g];
    float cxv = cxo[g];
    float chv = cho[g];
    float sgr = 1.f / (1.f + __expf(-rv));
    float z   = cxv + sgr * chv;
    float e2  = __expf(2.f * z);
    float th  = 1.f - 2.f / (e2 + 1.f);          // tanh(z)
    float sgu = 1.f / (1.f + __expf(-uv));
    float psv = ps[g];
    out[g] = psv * (1.f - sgu) + th * sgu;
  }
}

// ---------------- launch ----------------
extern "C" void kernel_launch(void* const* d_in, const int* in_sizes, int n_in,
                              void* d_out, int out_size, void* d_ws, size_t ws_size,
                              hipStream_t stream) {
  const float* x   = (const float*)d_in[0];
  const float* ps  = (const float*)d_in[1];
  const float* pi  = (const float*)d_in[2];
  const float* pps = (const float*)d_in[3];
  const float* mu  = (const float*)d_in[4];
  const float* mr  = (const float*)d_in[5];
  const float* mcx = (const float*)d_in[6];
  const float* mch = (const float*)d_in[7];
  const float* wr  = (const float*)d_in[8];
  const float* br  = (const float*)d_in[9];
  const float* wu  = (const float*)d_in[10];
  const float* bu  = (const float*)d_in[11];
  const float* wcx = (const float*)d_in[12];
  const float* bcx = (const float*)d_in[13];
  const float* wch = (const float*)d_in[14];
  const float* bch = (const float*)d_in[15];
  unsigned short* wp = (unsigned short*)d_ws;
  float* out = (float*)d_out;

  prep_weights<<<(WTOT + 255) / 256, 256, 0, stream>>>(wu, wr, wcx, wch, wp);

  dim3 grid(W_ / TILE_W, H_ / TILE_H, B_);       // 15 x 30 x 8
  dgru_main<<<grid, 256, 0, stream>>>(x, ps, pi, pps, mu, mr, mcx, mch,
                                      bu, br, bcx, bch, wp, out);
}
